// MultiHeadAttnBlock_11433202942699
// MI455X (gfx1250) — matrix operand order repaired and checked by reading, not compiled, not run on hardware
//
#include <hip/hip_runtime.h>

// MultiHeadAttnBlock for MI455X (gfx1250): f16 WMMA (16x16x32, f32 accum)
// for QKV/O projections and flash attention; f32 VALU for norms/softmax.
// gfx1250 features: async global->LDS (ASYNCcnt) with double-buffered tiles,
// DPP lane reductions, global prefetch.

typedef __attribute__((ext_vector_type(16))) _Float16 v16h;
typedef __attribute__((ext_vector_type(8)))  _Float16 v8h;
typedef __attribute__((ext_vector_type(8)))  float    v8f;

#define C_DIM  256
#define S_DIM  4096
#define BATCH  2
#define N_HEADS 4
#define ATT_D  64
#define GROUPS 32

// ---------------------------------------------------------------------------
// CDNA5 WMMA 16-bit fragment loaders (per ISA 7.12.2 register layouts).
// A (16x32, MxK): lane l holds row M=l&15; K chunks [c0,c0+8) and [c0+16,c0+24)
// with c0 = 8*(l>>4). Source row-major, K contiguous, 16B-aligned chunks.
__device__ __forceinline__ v16h load_frag_a(const _Float16* base, int stride_h, int lane) {
  const _Float16* r = base + (lane & 15) * stride_h + ((lane >> 4) & 1) * 8;
  v8h lo = *(const v8h*)(r);
  v8h hi = *(const v8h*)(r + 16);
  v16h a;
#pragma unroll
  for (int i = 0; i < 8; ++i) { a[i] = lo[i]; a[i + 8] = hi[i]; }
  return a;
}

// B (32x16, KxN): lane l holds column N=l&15; K chunks [c0,c0+8) and [c0+16,c0+24), c0 = 8*(l>>4).
// Source layout [n][k] row-major (k contiguous, stride_h halves per n).
__device__ __forceinline__ v16h load_frag_b(const _Float16* base, int stride_h, int lane) {
  const _Float16* r = base + (lane & 15) * stride_h + ((lane >> 4) & 1) * 8;
  v8h lo = *(const v8h*)(r);
  v8h hi = *(const v8h*)(r + 16);
  v16h b;
#pragma unroll
  for (int i = 0; i < 8; ++i) { b[i] = lo[i]; b[i + 8] = hi[i]; }
  return b;
}

// ---------------------------------------------------------------------------
// gfx1250 async global->LDS copy (GLOBAL_LOAD_ASYNC_TO_LDS_B128, ASYNCcnt).
__device__ __forceinline__ void async_copy_b128(const _Float16* gsrc, _Float16* lds_dst) {
  unsigned l = (unsigned)(size_t)lds_dst;               // low 32 bits = LDS offset
  unsigned long long g = (unsigned long long)(size_t)gsrc;
  asm volatile("global_load_async_to_lds_b128 %0, %1, off" :: "v"(l), "v"(g) : "memory");
}
__device__ __forceinline__ void wait_async0() {
  asm volatile("s_wait_asynccnt 0x0" ::: "memory");
}

// ---------------------------------------------------------------------------
// 16-lane xor reductions in pure VALU via DPP (no LDS bpermute round-trips).
template <int CTRL>
__device__ __forceinline__ float dpp_movf(float x) {
  return __int_as_float(__builtin_amdgcn_update_dpp(0, __float_as_int(x), CTRL, 0xf, 0xf, true));
}
__device__ __forceinline__ float red_max16(float v) {
  v = fmaxf(v, dpp_movf<0xB1>(v));    // quad_perm(1,0,3,2): xor 1
  v = fmaxf(v, dpp_movf<0x4E>(v));    // quad_perm(2,3,0,1): xor 2
  v = fmaxf(v, dpp_movf<0x141>(v));   // row_half_mirror: combine quads in 8
  v = fmaxf(v, dpp_movf<0x140>(v));   // row_mirror: combine 8-halves in 16
  return v;
}
__device__ __forceinline__ float red_sum16(float v) {
  v += dpp_movf<0xB1>(v);
  v += dpp_movf<0x4E>(v);
  v += dpp_movf<0x141>(v);
  v += dpp_movf<0x140>(v);
  return v;
}

// ---------------------------------------------------------------------------
// Kernel 1: GroupNorm for x (->hn) and y (->yn), f32 stats, f16 output.
__global__ void __launch_bounds__(256) gn16_kernel(
    const float* __restrict__ x, const float* __restrict__ y,
    const float* __restrict__ g1, const float* __restrict__ b1,
    const float* __restrict__ g2, const float* __restrict__ b2,
    _Float16* __restrict__ hn, _Float16* __restrict__ yn) {
  const int group = blockIdx.x, batch = blockIdx.y, sel = blockIdx.z;
  const float* src = sel ? y : x;
  const float* gam = sel ? g2 : g1;
  const float* bet = sel ? b2 : b1;
  _Float16* dst = sel ? yn : hn;
  const int tid = threadIdx.x;
  const int cpg = C_DIM / GROUPS;                       // 8
  const size_t base = ((size_t)batch * C_DIM + group * cpg) * S_DIM;
  const int n = cpg * S_DIM;                            // 32768
  float s1 = 0.f, s2 = 0.f;
  for (int i = tid; i < n; i += 256) {
    float v = src[base + i];
    s1 += v; s2 += v * v;
  }
  __shared__ float r1[256], r2[256];
  r1[tid] = s1; r2[tid] = s2;
  __syncthreads();
  for (int off = 128; off > 0; off >>= 1) {
    if (tid < off) { r1[tid] += r1[tid + off]; r2[tid] += r2[tid + off]; }
    __syncthreads();
  }
  const float mean = r1[0] / (float)n;
  const float var  = r2[0] / (float)n - mean * mean;
  const float rstd = rsqrtf(var + 1e-6f);
  for (int i = tid; i < n; i += 256) {
    int c = group * cpg + (i >> 12);
    float v = (src[base + i] - mean) * rstd * gam[c] + bet[c];
    dst[base + i] = (_Float16)v;
  }
}

// ---------------------------------------------------------------------------
// Kernel 2: weights f32 -> f16 (ATT^-0.5 folded into wq/bq).
__global__ void __launch_bounds__(256) wcvt_kernel(
    const float* __restrict__ wq, const float* __restrict__ wk,
    const float* __restrict__ wv, const float* __restrict__ wo,
    const float* __restrict__ bq, const float* __restrict__ bk,
    const float* __restrict__ bv, const float* __restrict__ bo,
    _Float16* __restrict__ wh, float* __restrict__ bh) {
  const int gid = blockIdx.x * 256 + threadIdx.x;       // 1024 blocks -> 262144
  const int sel = gid >> 16, idx = gid & 65535;
  const float* wsrc = (sel == 0) ? wq : (sel == 1) ? wk : (sel == 2) ? wv : wo;
  const float scale = (sel == 0) ? 0.125f : 1.0f;       // ATT^-0.5 = 1/8
  wh[gid] = (_Float16)(wsrc[idx] * scale);
  if (gid < 4 * C_DIM) {
    int s2 = gid >> 8, i2 = gid & 255;
    const float* bsrc = (s2 == 0) ? bq : (s2 == 1) ? bk : (s2 == 2) ? bv : bo;
    bh[gid] = bsrc[i2] * ((s2 == 0) ? 0.125f : 1.0f);
  }
}

// ---------------------------------------------------------------------------
// Kernel 3: WMMA GEMM  out[co][sp] = sum_ci W[co][ci] * X[ci][sp] + bias[co]
// Block tile 128(M) x 128(N), 8 waves of 32x64, K steps of 32, double-buffered
// async LDS staging (copy of step kk+1 overlaps compute of step kk).
// IN_MODE  0: X is f16 [batch][C][S]        1: X is f16 [batch][head][S][64]
// OUT_MODE 0: f16 [batch][C][S]   1: f16 [batch][head][S][64]
//          2: f32 [batch][C][S] + residual
template<int IN_MODE, int OUT_MODE>
__global__ void __launch_bounds__(256) gemm256_kernel(
    const _Float16* __restrict__ W, const float* __restrict__ bias,
    const _Float16* __restrict__ X, const float* __restrict__ resid,
    _Float16* __restrict__ outh, float* __restrict__ outf) {
  const int batch = blockIdx.z;
  const int m_base = blockIdx.y * 128;
  const int n_base = blockIdx.x * 128;
  const int tid = threadIdx.x;
  const int lane = tid & 31;
  const int wid = tid >> 5;
  const int wm = (wid & 3) * 32;                        // 4 waves along M
  const int wn = (wid >> 2) * 64;                       // 2 waves along N

  __shared__ __align__(16) _Float16 sW[2][128 * 40];    // [m][k], pad to 40
  __shared__ __align__(16) _Float16 sX[2][128 * 40];    // [n][k], pad to 40

  auto stage = [&](int kk, int bufi) {
    const int k0 = kk * 32;
    _Float16* wdst = &sW[bufi][0];
    _Float16* xdst = &sX[bufi][0];
    // W tile [128 m][32 k]: async b128 direct global->LDS
    for (int i = tid; i < 512; i += 256) {
      int m = i >> 2, c8 = (i & 3) * 8;
      async_copy_b128(W + (m_base + m) * C_DIM + k0 + c8, wdst + m * 40 + c8);
    }
    if (IN_MODE == 0) {
      // transpose from [c][s]: scalar gather/scatter (coalesced along n)
      for (int i = tid; i < 128 * 32; i += 256) {
        int k = i >> 7, n = i & 127;
        xdst[n * 40 + k] = X[((size_t)batch * C_DIM + k0 + k) * S_DIM + n_base + n];
      }
    } else {
      // [b][h][s][d]: k (=d) already contiguous -> async b128
      for (int i = tid; i < 512; i += 256) {
        int n = i >> 2, c8 = (i & 3) * 8;
        int ci = k0 + c8;
        async_copy_b128(
            X + (((size_t)batch * N_HEADS + (ci >> 6)) * S_DIM + n_base + n) * ATT_D + (ci & 63),
            xdst + n * 40 + c8);
      }
    }
  };

  v8f acc[2][4] = {};
  stage(0, 0);

  for (int kk = 0; kk < 8; ++kk) {
    const int cur = kk & 1;
    wait_async0();                                      // own copies of step kk done
    __syncthreads();                                    // everyone's copies + prev reads done
    if (kk < 7) stage(kk + 1, cur ^ 1);                 // overlap next copy with compute

    const _Float16* ws = &sW[cur][0];
    const _Float16* xs = &sX[cur][0];
    v16h a[2], b[4];
#pragma unroll
    for (int m = 0; m < 2; ++m) a[m] = load_frag_a(ws + (wm + m * 16) * 40, 40, lane);
#pragma unroll
    for (int n = 0; n < 4; ++n) b[n] = load_frag_b(xs + (wn + n * 16) * 40, 40, lane);
#pragma unroll
    for (int m = 0; m < 2; ++m)
#pragma unroll
      for (int n = 0; n < 4; ++n)
        acc[m][n] = __builtin_amdgcn_wmma_f32_16x16x32_f16(
            false, a[m], false, b[n], (short)0, acc[m][n], false, false);
  }

  // C layout: lane l, reg r -> M = r + 8*(l>>4), N = l&15
  const int rl = 8 * (lane >> 4);
  const int cl = lane & 15;
#pragma unroll
  for (int m = 0; m < 2; ++m)
#pragma unroll
    for (int n = 0; n < 4; ++n)
#pragma unroll
      for (int r = 0; r < 8; ++r) {
        int co = m_base + wm + m * 16 + rl + r;
        int sp = n_base + wn + n * 16 + cl;
        float v = acc[m][n][r] + bias[co];
        if (OUT_MODE == 0) {
          outh[((size_t)batch * C_DIM + co) * S_DIM + sp] = (_Float16)v;
        } else if (OUT_MODE == 1) {
          outh[(((size_t)batch * N_HEADS + (co >> 6)) * S_DIM + sp) * ATT_D + (co & 63)] =
              (_Float16)v;
        } else {
          size_t id = ((size_t)batch * C_DIM + co) * S_DIM + sp;
          outf[id] = v + resid[id];
        }
      }
}

// ---------------------------------------------------------------------------
// Kernel 4: flash attention per (batch, head). 4 waves x 16 queries per block.
// qt and kt are [b][h][s][d] (Q/K fragments read straight from global);
// vm is [b][c][s], staged to LDS [d][key] via double-buffered async copies.
__global__ void __launch_bounds__(128) attn64_kernel(
    const _Float16* __restrict__ qt, const _Float16* __restrict__ kt,
    const _Float16* __restrict__ vm, _Float16* __restrict__ ao) {
  const int qtile = blockIdx.x;
  const int head  = blockIdx.y;
  const int batch = blockIdx.z;
  const int tid = threadIdx.x;
  const int lane = tid & 31;
  const int wid = tid >> 5;                             // 0..3

  const _Float16* qbase = qt + (((size_t)batch * N_HEADS + head) * S_DIM) * ATT_D;
  const _Float16* kbase = kt + (((size_t)batch * N_HEADS + head) * S_DIM) * ATT_D;
  const _Float16* vbase = vm + ((size_t)batch * C_DIM + head * ATT_D) * S_DIM;

  __shared__ __align__(16) _Float16 Vt[2][64 * 72];     // [d][key], double-buffered
  __shared__ __align__(16) _Float16 Pb[4][16 * 72];     // per-wave P [q][key]

  auto stage_v = [&](int kb, int bufi) {
    const int key0 = kb * 64;
    _Float16* vdst = &Vt[bufi][0];
    for (int i = tid; i < 512; i += 128) {
      int d = i >> 3, c8 = (i & 7) * 8;
      async_copy_b128(vbase + (size_t)d * S_DIM + key0 + c8, vdst + d * 72 + c8);
    }
  };

  const int q0 = qtile * 64 + wid * 16;
  // Q strip [16 q][64 d] straight from global (row-major, 16B chunks per lane)
  v16h qa0 = load_frag_a(qbase + (size_t)q0 * ATT_D, ATT_D, lane);
  v16h qa1 = load_frag_a(qbase + (size_t)q0 * ATT_D + 32, ATT_D, lane);

  v8f o[4] = {};
  float mrow[8], lrow[8];
#pragma unroll
  for (int r = 0; r < 8; ++r) { mrow[r] = -1e30f; lrow[r] = 0.f; }

  stage_v(0, 0);

  for (int kb = 0; kb < S_DIM / 64; ++kb) {
    const int key0 = kb * 64;
    const int cur = kb & 1;
    wait_async0();
    __syncthreads();
    if (kb < S_DIM / 64 - 1) {
      stage_v(kb + 1, cur ^ 1);                         // V copy overlaps compute
      // prefetch next K block (consumed directly from global next iteration)
      __builtin_prefetch(kbase + (size_t)(key0 + 64) * ATT_D + tid * 64, 0, 0);
    }
    const _Float16* vs = &Vt[cur][0];

    // Preload all K B-frags (one load clause), then S = Q K^T
    v16h kf[8];
#pragma unroll
    for (int n = 0; n < 4; ++n) {
      const _Float16* kblk = kbase + (size_t)(key0 + n * 16) * ATT_D;
      kf[2 * n]     = load_frag_b(kblk, ATT_D, lane);       // d 0..31
      kf[2 * n + 1] = load_frag_b(kblk + 32, ATT_D, lane);  // d 32..63
    }
    v8f s[4] = {};
#pragma unroll
    for (int n = 0; n < 4; ++n) {
      s[n] = __builtin_amdgcn_wmma_f32_16x16x32_f16(false, qa0, false, kf[2 * n], (short)0, s[n], false, false);
      s[n] = __builtin_amdgcn_wmma_f32_16x16x32_f16(false, qa1, false, kf[2 * n + 1], (short)0, s[n], false, false);
    }

    // online softmax; each row's 16 columns live across one 16-lane half
    float pscale[8], rsum[8];
#pragma unroll
    for (int r = 0; r < 8; ++r) {
      float rm = fmaxf(fmaxf(s[0][r], s[1][r]), fmaxf(s[2][r], s[3][r]));
      rm = red_max16(rm);
      float mn = fmaxf(mrow[r], rm);
      pscale[r] = __expf(mrow[r] - mn);
      mrow[r] = mn;
      rsum[r] = 0.f;
    }
#pragma unroll
    for (int n = 0; n < 4; ++n)
#pragma unroll
      for (int r = 0; r < 8; ++r) {
        float p = __expf(s[n][r] - mrow[r]);
        s[n][r] = p;
        rsum[r] += p;
      }
#pragma unroll
    for (int r = 0; r < 8; ++r) {
      lrow[r] = lrow[r] * pscale[r] + red_sum16(rsum[r]);
    }
#pragma unroll
    for (int d = 0; d < 4; ++d)
#pragma unroll
      for (int r = 0; r < 8; ++r) o[d][r] *= pscale[r];

    // P: C-layout regs -> LDS -> A fragments (same-wave DS RAW: s_wait_dscnt)
    const int rl = 8 * (lane >> 4), cl = lane & 15;
    _Float16* pb = &Pb[wid][0];
#pragma unroll
    for (int n = 0; n < 4; ++n)
#pragma unroll
      for (int r = 0; r < 8; ++r)
        pb[(rl + r) * 72 + n * 16 + cl] = (_Float16)s[n][r];
    asm volatile("s_wait_dscnt 0x0" ::: "memory");
    v16h pa0 = load_frag_a(pb, 72, lane);
    v16h pa1 = load_frag_a(pb + 32, 72, lane);

    // O += P V : 4 d-subtiles x (K=64 keys -> 2 WMMA)
#pragma unroll
    for (int d = 0; d < 4; ++d) {
      v16h vb0 = load_frag_b(vs + d * 16 * 72, 72, lane);
      v16h vb1 = load_frag_b(vs + d * 16 * 72 + 32, 72, lane);
      o[d] = __builtin_amdgcn_wmma_f32_16x16x32_f16(false, pa0, false, vb0, (short)0, o[d], false, false);
      o[d] = __builtin_amdgcn_wmma_f32_16x16x32_f16(false, pa1, false, vb1, (short)0, o[d], false, false);
    }
  }

  const int rl = 8 * (lane >> 4), cl = lane & 15;
#pragma unroll
  for (int r = 0; r < 8; ++r) {
    float inv = 1.0f / lrow[r];
    int q = q0 + rl + r;
#pragma unroll
    for (int d = 0; d < 4; ++d) {
      int dd = d * 16 + cl;
      ao[(((size_t)batch * N_HEADS + head) * S_DIM + q) * ATT_D + dd] =
          (_Float16)(o[d][r] * inv);
    }
  }
}

// ---------------------------------------------------------------------------
extern "C" void kernel_launch(void* const* d_in, const int* in_sizes, int n_in,
                              void* d_out, int out_size, void* d_ws, size_t ws_size,
                              hipStream_t stream) {
  (void)in_sizes; (void)n_in; (void)out_size; (void)ws_size;
  const float* x  = (const float*)d_in[0];
  const float* y  = (const float*)d_in[1];
  const float* g1 = (const float*)d_in[2];
  const float* b1 = (const float*)d_in[3];
  const float* g2 = (const float*)d_in[4];
  const float* b2 = (const float*)d_in[5];
  const float* wq = (const float*)d_in[6];
  const float* bq = (const float*)d_in[7];
  const float* wk = (const float*)d_in[8];
  const float* bk = (const float*)d_in[9];
  const float* wv = (const float*)d_in[10];
  const float* bv = (const float*)d_in[11];
  const float* wo = (const float*)d_in[12];
  const float* bo = (const float*)d_in[13];
  float* out = (float*)d_out;

  // workspace layout (~24.5 MB): all buffers fully overwritten each call
  const size_t TENS = (size_t)BATCH * C_DIM * S_DIM;    // 2,097,152 halves
  _Float16* hn = (_Float16*)d_ws;
  _Float16* yn = hn + TENS;
  _Float16* wh = yn + TENS;                             // 4*256*256 halves
  float*    bh = (float*)(wh + 4 * C_DIM * C_DIM);      // 4*256 floats
  _Float16* qt = (_Float16*)(bh + 4 * C_DIM);
  _Float16* kt = qt + TENS;
  _Float16* vm = kt + TENS;
  _Float16* ao = vm + TENS;

  gn16_kernel<<<dim3(GROUPS, BATCH, 2), 256, 0, stream>>>(x, y, g1, b1, g2, b2, hn, yn);
  wcvt_kernel<<<1024, 256, 0, stream>>>(wq, wk, wv, wo, bq, bk, bv, bo, wh, bh);

  const dim3 ggrid(S_DIM / 128, C_DIM / 128, BATCH);
  // Q = wq * yn  -> [b][h][s][d] (scale folded into weights)
  gemm256_kernel<0, 1><<<ggrid, 256, 0, stream>>>(wh, bh, yn, nullptr, qt, nullptr);
  // K = wk * hn  -> [b][h][s][d]
  gemm256_kernel<0, 1><<<ggrid, 256, 0, stream>>>(wh + 65536, bh + 256, hn, nullptr, kt, nullptr);
  // V = wv * hn  -> [b][c][s]
  gemm256_kernel<0, 0><<<ggrid, 256, 0, stream>>>(wh + 131072, bh + 512, hn, nullptr, vm, nullptr);

  attn64_kernel<<<dim3(S_DIM / 64, N_HEADS, BATCH), 128, 0, stream>>>(qt, kt, vm, ao);

  // out = wo * attn + bo + x  (f32)
  gemm256_kernel<1, 2><<<ggrid, 256, 0, stream>>>(wh + 196608, bh + 768, ao, x, nullptr, out);
}
